// GCN_71725953844015
// MI455X (gfx1250) — matrix operand order repaired and checked by reading, not compiled, not run on hardware
//
#include <hip/hip_runtime.h>
#include <hip/hip_bf16.h>

// ---------------------------------------------------------------------------
// 2-layer GCN for MI455X (gfx1250, wave32, WMMA).
//   layer: t = X @ W (WMMA f16->f32), agg[dst] += t[src]*dinv[src]*dinv[dst]
//          (edge scatter via global_atomic_add_f32), + self-loop t[i]*dinv[i]^2,
//          + bias, (ReLU).
// ---------------------------------------------------------------------------

typedef __attribute__((ext_vector_type(16))) _Float16 v16h;
typedef __attribute__((ext_vector_type(8)))  _Float16 v8h;
typedef __attribute__((ext_vector_type(8)))  float    v8f;

#define NN   100000
#define NE   1600000
#define INC  128
#define HID  64
#define OUTC 32

// ---- A-fragment (16-bit A, 16x32 MxK, ISA 7.12.2) --------------------------
// lane m=lane&15 holds row M=m; hi=lane>>4.
// elements 0..15 <-> K = hi*16 + 0..15 (contiguous, matches B-fragment)
__device__ __forceinline__ v16h load_afrag(const _Float16* __restrict__ rowk0, int hi) {
    return *reinterpret_cast<const v16h*>(rowk0 + hi * 16);
}

// ---- B-fragment (16-bit B, 32x16 KxN) --------------------------------------
// lane n=lane&15 holds column N=n; lanes 0-15: K=0..15, lanes 16-31: K=16..31
// -> contiguous 16 f16 from an N-major (transposed) weight copy.
__device__ __forceinline__ v16h load_bfrag(const _Float16* __restrict__ colk0, int hi) {
    return *reinterpret_cast<const v16h*>(colk0 + hi * 16);
}

// ---- GEMM1: t1[NN,64] = xh[NN,128] @ W1 ------------------------------------
__global__ __launch_bounds__(256) void gcn_gemm1_wmma(
    const _Float16* __restrict__ xh, const _Float16* __restrict__ w1t,
    float* __restrict__ t1, int mtiles) {
    const int lane  = threadIdx.x & 31;
    const int wave  = threadIdx.x >> 5;
    const int tileM = blockIdx.x * (blockDim.x >> 5) + wave;   // wave-uniform
    if (tileM >= mtiles) return;                               // whole wave exits
    const int m  = lane & 15;
    const int hi = lane >> 4;

    v8f c0 = {}, c1 = {}, c2 = {}, c3 = {};
    const _Float16* arow = xh + (size_t)(tileM * 16 + m) * INC;
#pragma unroll
    for (int kt = 0; kt < 4; ++kt) {                           // K = 128 = 4*32
        const v16h a  = load_afrag(arow + kt * 32, hi);
        const v16h b0 = load_bfrag(w1t + (size_t)(0 * 16 + m) * INC + kt * 32, hi);
        const v16h b1 = load_bfrag(w1t + (size_t)(1 * 16 + m) * INC + kt * 32, hi);
        const v16h b2 = load_bfrag(w1t + (size_t)(2 * 16 + m) * INC + kt * 32, hi);
        const v16h b3 = load_bfrag(w1t + (size_t)(3 * 16 + m) * INC + kt * 32, hi);
        c0 = __builtin_amdgcn_wmma_f32_16x16x32_f16(false, a, false, b0, (short)0, c0, false, false);
        c1 = __builtin_amdgcn_wmma_f32_16x16x32_f16(false, a, false, b1, (short)0, c1, false, false);
        c2 = __builtin_amdgcn_wmma_f32_16x16x32_f16(false, a, false, b2, (short)0, c2, false, false);
        c3 = __builtin_amdgcn_wmma_f32_16x16x32_f16(false, a, false, b3, (short)0, c3, false, false);
    }
    // C/D layout: N = lane&15, M = hi*8 + r
    float* crow = t1 + (size_t)(tileM * 16 + hi * 8) * HID + m;
#pragma unroll
    for (int r = 0; r < 8; ++r) {
        crow[r * HID +  0] = c0[r];
        crow[r * HID + 16] = c1[r];
        crow[r * HID + 32] = c2[r];
        crow[r * HID + 48] = c3[r];
    }
}

// ---- GEMM2: t2[NN,32] = h1h[NN,64] @ W2 ------------------------------------
__global__ __launch_bounds__(256) void gcn_gemm2_wmma(
    const _Float16* __restrict__ hh, const _Float16* __restrict__ w2t,
    float* __restrict__ t2, int mtiles) {
    const int lane  = threadIdx.x & 31;
    const int wave  = threadIdx.x >> 5;
    const int tileM = blockIdx.x * (blockDim.x >> 5) + wave;
    if (tileM >= mtiles) return;
    const int m  = lane & 15;
    const int hi = lane >> 4;

    v8f c0 = {}, c1 = {};
    const _Float16* arow = hh + (size_t)(tileM * 16 + m) * HID;
#pragma unroll
    for (int kt = 0; kt < 2; ++kt) {                           // K = 64 = 2*32
        const v16h a  = load_afrag(arow + kt * 32, hi);
        const v16h b0 = load_bfrag(w2t + (size_t)(0 * 16 + m) * HID + kt * 32, hi);
        const v16h b1 = load_bfrag(w2t + (size_t)(1 * 16 + m) * HID + kt * 32, hi);
        c0 = __builtin_amdgcn_wmma_f32_16x16x32_f16(false, a, false, b0, (short)0, c0, false, false);
        c1 = __builtin_amdgcn_wmma_f32_16x16x32_f16(false, a, false, b1, (short)0, c1, false, false);
    }
    float* crow = t2 + (size_t)(tileM * 16 + hi * 8) * OUTC + m;
#pragma unroll
    for (int r = 0; r < 8; ++r) {
        crow[r * OUTC +  0] = c0[r];
        crow[r * OUTC + 16] = c1[r];
    }
}

// ---- elementwise / graph kernels -------------------------------------------
__global__ void gcn_fill_f32(float* __restrict__ p, float v, int n) {
    int i = blockIdx.x * blockDim.x + threadIdx.x;
    if (i < n) p[i] = v;
}

__global__ void gcn_f32_to_f16(const float* __restrict__ in, _Float16* __restrict__ out, int n) {
    int i = blockIdx.x * blockDim.x + threadIdx.x;
    if (i < n) out[i] = (_Float16)in[i];
}

// Wt[n*K + k] = W[k*N + n]  (N-major f16 copy for B-fragments)
__global__ void gcn_transpose_f16(const float* __restrict__ W, _Float16* __restrict__ Wt,
                                  int K, int N) {
    int i = blockIdx.x * blockDim.x + threadIdx.x;
    if (i < K * N) {
        int n = i / K, k = i - n * K;
        Wt[i] = (_Float16)W[k * N + n];
    }
}

__global__ void gcn_deg_count(const int* __restrict__ dst, float* __restrict__ deg, int ne) {
    int i = blockIdx.x * blockDim.x + threadIdx.x;
    if (i < ne) atomicAdd(&deg[dst[i]], 1.0f);
}

__global__ void gcn_make_dinv(const float* __restrict__ deg, float* __restrict__ dinv, int n) {
    int i = blockIdx.x * blockDim.x + threadIdx.x;
    if (i < n) dinv[i] = __frsqrt_rn(deg[i]);   // deg >= 1 (self-loop)
}

// layer-1 edge scatter: 16 lanes/edge, 4 channels each (HID=64)
__global__ void gcn_scatter1(const int* __restrict__ ei, const float* __restrict__ t1,
                             const float* __restrict__ dinv, float* __restrict__ agg, int ne) {
    int tid  = blockIdx.x * blockDim.x + threadIdx.x;
    int edge = tid >> 4;
    if (edge >= ne) return;
    int sub = tid & 15;
    int src = ei[edge];
    int dst = ei[NE + edge];
    float norm = dinv[src] * dinv[dst];
    float4 v = *reinterpret_cast<const float4*>(t1 + (size_t)src * HID + sub * 4);
    float* o = agg + (size_t)dst * HID + sub * 4;
    atomicAdd(o + 0, v.x * norm);
    atomicAdd(o + 1, v.y * norm);
    atomicAdd(o + 2, v.z * norm);
    atomicAdd(o + 3, v.w * norm);
}

// agg + self-loop + bias, ReLU, cast to f16 for layer-2 GEMM
__global__ void gcn_finalize1(const float* __restrict__ agg, const float* __restrict__ t1,
                              const float* __restrict__ dinv, const float* __restrict__ b1,
                              _Float16* __restrict__ hh, int total) {
    int i = blockIdx.x * blockDim.x + threadIdx.x;
    if (i >= total) return;
    int node = i >> 6;           // /HID
    int ch   = i & (HID - 1);
    float di = dinv[node];
    float v  = agg[i] + t1[i] * di * di + b1[ch];
    hh[i] = (_Float16)fmaxf(v, 0.0f);
}

// layer-2 edge scatter: 8 lanes/edge, 4 channels each (OUTC=32), into d_out
__global__ void gcn_scatter2(const int* __restrict__ ei, const float* __restrict__ t2,
                             const float* __restrict__ dinv, float* __restrict__ out, int ne) {
    int tid  = blockIdx.x * blockDim.x + threadIdx.x;
    int edge = tid >> 3;
    if (edge >= ne) return;
    int sub = tid & 7;
    int src = ei[edge];
    int dst = ei[NE + edge];
    float norm = dinv[src] * dinv[dst];
    float4 v = *reinterpret_cast<const float4*>(t2 + (size_t)src * OUTC + sub * 4);
    float* o = out + (size_t)dst * OUTC + sub * 4;
    atomicAdd(o + 0, v.x * norm);
    atomicAdd(o + 1, v.y * norm);
    atomicAdd(o + 2, v.z * norm);
    atomicAdd(o + 3, v.w * norm);
}

__global__ void gcn_finalize2(float* __restrict__ out, const float* __restrict__ t2,
                              const float* __restrict__ dinv, const float* __restrict__ b2,
                              int total) {
    int i = blockIdx.x * blockDim.x + threadIdx.x;
    if (i >= total) return;
    int node = i >> 5;           // /OUTC
    int ch   = i & (OUTC - 1);
    float di = dinv[node];
    out[i] = out[i] + t2[i] * di * di + b2[ch];
}

// ---------------------------------------------------------------------------
extern "C" void kernel_launch(void* const* d_in, const int* in_sizes, int n_in,
                              void* d_out, int out_size, void* d_ws, size_t ws_size,
                              hipStream_t stream) {
    const float* x  = (const float*)d_in[0];
    const int*   ei = (const int*)d_in[1];     // edge_index [2, NE]
    const float* W1 = (const float*)d_in[2];
    const float* b1 = (const float*)d_in[3];
    const float* W2 = (const float*)d_in[4];
    const float* b2 = (const float*)d_in[5];
    float* out = (float*)d_out;

    // workspace carve-out (256B-aligned slices)
    char* ws = (char*)d_ws;
    size_t off = 0;
    auto take = [&](size_t bytes) -> char* {
        char* p = ws + off;
        off = (off + bytes + 255) & ~(size_t)255;
        return p;
    };
    _Float16* xh   = (_Float16*)take((size_t)NN * INC * sizeof(_Float16));
    _Float16* w1t  = (_Float16*)take((size_t)HID * INC * sizeof(_Float16));
    _Float16* w2t  = (_Float16*)take((size_t)OUTC * HID * sizeof(_Float16));
    float*    t1   = (float*)take((size_t)NN * HID * sizeof(float));
    float*    agg1 = (float*)take((size_t)NN * HID * sizeof(float));
    _Float16* h1h  = (_Float16*)take((size_t)NN * HID * sizeof(_Float16));
    float*    t2   = (float*)take((size_t)NN * OUTC * sizeof(float));
    float*    deg  = (float*)take((size_t)NN * sizeof(float));
    float*    dinv = (float*)take((size_t)NN * sizeof(float));

    const int T = 256;
    auto blk = [](int n, int t) { return (n + t - 1) / t; };

    // normalization: deg = 1 (self-loop) + in-degree; dinv = rsqrt(deg)
    gcn_fill_f32<<<blk(NN, T), T, 0, stream>>>(deg, 1.0f, NN);
    gcn_deg_count<<<blk(NE, T), T, 0, stream>>>(ei + NE, deg, NE);
    gcn_make_dinv<<<blk(NN, T), T, 0, stream>>>(deg, dinv, NN);

    // fp16 operand staging
    gcn_f32_to_f16<<<blk(NN * INC, T), T, 0, stream>>>(x, xh, NN * INC);
    gcn_transpose_f16<<<blk(INC * HID, T), T, 0, stream>>>(W1, w1t, INC, HID);
    gcn_transpose_f16<<<blk(HID * OUTC, T), T, 0, stream>>>(W2, w2t, HID, OUTC);

    // layer 1
    const int mtiles = NN / 16;                  // 6250
    const int wavesPerBlock = T / 32;            // 8
    gcn_gemm1_wmma<<<blk(mtiles, wavesPerBlock), T, 0, stream>>>(xh, w1t, t1, mtiles);
    gcn_fill_f32<<<blk(NN * HID, T), T, 0, stream>>>(agg1, 0.0f, NN * HID);
    gcn_scatter1<<<blk(NE * 16, T), T, 0, stream>>>(ei, t1, dinv, agg1, NE);
    gcn_finalize1<<<blk(NN * HID, T), T, 0, stream>>>(agg1, t1, dinv, b1, h1h, NN * HID);

    // layer 2
    gcn_gemm2_wmma<<<blk(mtiles, wavesPerBlock), T, 0, stream>>>(h1h, w2t, t2, mtiles);
    gcn_fill_f32<<<blk(NN * OUTC, T), T, 0, stream>>>(out, 0.0f, NN * OUTC);
    gcn_scatter2<<<blk(NE * 8, T), T, 0, stream>>>(ei, t2, dinv, out, NE);
    gcn_finalize2<<<blk(NN * OUTC, T), T, 0, stream>>>(out, t2, dinv, b2, NN * OUTC);
}
